// WeatherGNNEncoder_25220047962307
// MI455X (gfx1250) — matrix operand order repaired and checked by reading, not compiled, not run on hardware
//
#include <hip/hip_runtime.h>
#include <hip/hip_bf16.h>
#include <stdint.h>

// ---------------- problem constants ----------------
#define GH 721
#define GW 1440
#define C_IN 78
#define C 86           // C_IN + 8 additional
#define CP 96          // padded K for GEMM1 (3 x 32)
#define DM 256         // hidden/output width
#define NROWS (5882 * 128)
#define RB 64          // rows per block
#define NBLK (NROWS / RB)   // 11764
#define PERSIST 1024   // persistent workgroups

#define HFS 264        // fp32 hidden row stride (pad)
#define HBS 272        // bf16 hidden row stride (pad, 16B aligned: 272*2=544)

// workspace layout (dwords)
#define W1F_DW (3 * 16 * 32 * 8)    // 12288
#define W2F_DW (8 * 16 * 32 * 8)    // 32768
#define WF_DW (W1F_DW + W2F_DW)     // 45056
#define B1F_DW_OFF WF_DW

// LDS byte offsets
#define SM_WSM   0        // 180224 B : swizzled W1|W2 fragments
#define SM_ABF   180224   // 12288  B : bf16 A tiles 64x96
#define SM_HF    192512   // 67584  B : fp32 hidden 64x264
#define SM_HBF   260096   // 34816  B : bf16 hidden 64x272
#define SM_LNS   294912   // 1024 B
#define SM_LNO   295936   // 1024 B
#define SM_B1    296960   // 1024 B
#define SM_B2    297984   // 1024 B
#define SM_TOTAL 299008

typedef __attribute__((ext_vector_type(16))) __bf16 v16bf;
typedef __attribute__((ext_vector_type(8)))  float  v8f;
typedef __attribute__((ext_vector_type(4)))  int    v4i;

union Frag {
    v16bf v;
    uint4 q[2];
};

__device__ __forceinline__ unsigned short f2bf(float f) {
    uint32_t u = __float_as_uint(f);
    uint32_t r = (u + 0x7FFFu + ((u >> 16) & 1u)) >> 16;
    return (unsigned short)r;
}

#define HAS_ASYNC_LDS (__has_builtin(__builtin_amdgcn_global_load_async_to_lds_b128) && \
                       __has_builtin(__builtin_amdgcn_s_wait_asynccnt))

// ------------- prep: folded bias b1' = b1 - sum_c mean[c]*inv_std[c]*W1[c][d] -------------
__global__ void prep_b1(const float* __restrict__ means, const float* __restrict__ stds,
                        const float* __restrict__ W1, const float* __restrict__ b1,
                        float* __restrict__ b1f) {
    int d = threadIdx.x;
    float acc = b1[d];
    for (int c = 0; c < C; ++c) {
        float is = 1.0f / (stds[c] + 1e-7f);
        acc -= means[c] * is * W1[c * DM + d];
    }
    b1f[d] = acc;
}

// ------------- prep: W1' = W1 * inv_std, bf16, swizzled into WMMA B-fragment order -------------
// frag dword index t -> j = t&7, lane = (t>>3)&31, ct = (t>>8)&15, kt = t>>12
// B layout (32x16): lanes 0-15 hold K = kt*32 + {2j,2j+1}, col n = lane; lanes 16-31: K += 16.
__global__ void prep_w1frag(const float* __restrict__ stds, const float* __restrict__ W1,
                            uint32_t* __restrict__ w1f) {
    int t = blockIdx.x * blockDim.x + threadIdx.x;
    if (t >= W1F_DW) return;
    int j = t & 7, lane = (t >> 3) & 31, ct = (t >> 8) & 15, kt = t >> 12;
    int n = ct * 16 + (lane & 15);
    int k0 = kt * 32 + ((lane >> 4) << 4) + 2 * j;
    float v0 = 0.f, v1 = 0.f;
    if (k0 < C)     v0 = W1[k0 * DM + n]       / (stds[k0] + 1e-7f);
    if (k0 + 1 < C) v1 = W1[(k0 + 1) * DM + n] / (stds[k0 + 1] + 1e-7f);
    w1f[t] = (uint32_t)f2bf(v0) | ((uint32_t)f2bf(v1) << 16);
}

__global__ void prep_w2frag(const float* __restrict__ W2, uint32_t* __restrict__ w2f) {
    int t = blockIdx.x * blockDim.x + threadIdx.x;
    if (t >= W2F_DW) return;
    int j = t & 7, lane = (t >> 3) & 31, ct = (t >> 8) & 15, kt = t >> 12;
    int n = ct * 16 + (lane & 15);
    int k0 = kt * 32 + ((lane >> 4) << 4) + 2 * j;
    float v0 = W2[k0 * DM + n];
    float v1 = W2[(k0 + 1) * DM + n];
    w2f[t] = (uint32_t)f2bf(v0) | ((uint32_t)f2bf(v1) << 16);
}

// ---------------- fused persistent main kernel ----------------
__global__ void __launch_bounds__(256)
weather_encoder_main(const float* __restrict__ var_grid,
                     const int* __restrict__ nearby_idx,
                     const uint32_t* __restrict__ wfrag,      // W1F | W2F contiguous
                     const float* __restrict__ b1f,
                     const float* __restrict__ ln_scale,
                     const float* __restrict__ ln_offset,
                     const float* __restrict__ b2,
                     float* __restrict__ out) {
    extern __shared__ __align__(16) char smem[];
    uint32_t*       Wsm  = (uint32_t*)(smem + SM_WSM);
    unsigned short* Abf  = (unsigned short*)(smem + SM_ABF);
    float*          Hf   = (float*)(smem + SM_HF);
    unsigned short* Hbf  = (unsigned short*)(smem + SM_HBF);
    float*          lnSs = (float*)(smem + SM_LNS);
    float*          lnOs = (float*)(smem + SM_LNO);
    float*          b1s  = (float*)(smem + SM_B1);
    float*          b2s  = (float*)(smem + SM_B2);

    const int tid = threadIdx.x;

    // ---- one-time staging (persistent workgroup) ----
    lnSs[tid] = ln_scale[tid];
    lnOs[tid] = ln_offset[tid];
    b1s[tid]  = b1f[tid];
    b2s[tid]  = b2[tid];

#if HAS_ASYNC_LDS
    {
        #pragma unroll 4
        for (int i = tid; i < WF_DW / 4; i += 256) {
            __builtin_amdgcn_global_load_async_to_lds_b128(
                (__attribute__((address_space(1))) v4i*)(wfrag + 4 * i),
                (__attribute__((address_space(3))) v4i*)((uint4*)Wsm + i),
                0, 0);
        }
        __builtin_amdgcn_s_wait_asynccnt(0);
    }
#else
    {
        const uint4* src = (const uint4*)wfrag;
        uint4* dst = (uint4*)Wsm;
        #pragma unroll 4
        for (int i = tid; i < WF_DW / 4; i += 256) dst[i] = src[i];
    }
#endif
    __syncthreads();

    const int wv   = tid >> 5;
    const int lane = tid & 31;
    const int rtg  = wv >> 2;          // row tile group: rows tiles {2*rtg, 2*rtg+1}
    const int ctg  = wv & 3;           // col tile group: col tiles {4*ctg .. 4*ctg+3}
    const int m    = lane & 15;
    const int half = lane >> 4;
    const int p4   = tid & 3;          // 4 lanes per row (gather / LN mapping)
    const int r4   = tid >> 2;

    for (int blk = blockIdx.x; blk < NBLK; blk += gridDim.x) {
        // ---- gather 64 rows + analytic channels, bf16 into A tiles ----
        int gnext = -1;
        {
            int g = nearby_idx[blk * RB + r4];
            int nb = blk + gridDim.x;
            if (nb < NBLK) gnext = nearby_idx[nb * RB + r4];
            const float* src = var_grid + (size_t)g * C_IN;   // 8B aligned (g*312)
            unsigned short* dst = Abf + r4 * CP;
            #pragma unroll
            for (int s = p4; s < C_IN / 2; s += 4) {          // 39 float2 slots
                float2 f = *(const float2*)(src + 2 * s);
                dst[2 * s]     = f2bf(f.x);
                dst[2 * s + 1] = f2bf(f.y);
            }
            if (p4 == 0) {
                int hh = g / GW, ww = g - hh * GW;
                const float D2R = 0.017453292519943295f;
                float lat = (-90.0f + (float)hh * (180.0f / 720.0f)) * D2R;
                float lon = ((float)ww * (360.0f / 1439.0f)) * D2R;
                dst[78] = f2bf(__sinf(lat));
                dst[79] = f2bf(__cosf(lat));
                dst[80] = f2bf(__sinf(lon));
                dst[81] = f2bf(__cosf(lon));
                dst[82] = f2bf(0.5f);                         // 182.5/365
                #pragma unroll
                for (int c = 83; c < CP; ++c) dst[c] = 0;     // zero pads + K padding
            }
        }
        __syncthreads();

        // ---- GEMM1: hidden = relu(A @ W1' + b1'); 2 row-tiles x 4 col-tiles per wave ----
        {
            const unsigned short* ar0 = Abf + ((rtg * 2 + 0) * 16 + m) * CP;
            const unsigned short* ar1 = Abf + ((rtg * 2 + 1) * 16 + m) * CP;
            v8f acc[2][4] = {};
            #pragma unroll
            for (int kt = 0; kt < 3; ++kt) {
                int ko = kt * 32 + half * 16;
                Frag A0, A1;
                A0.q[0] = *(const uint4*)(ar0 + ko);
                A0.q[1] = *(const uint4*)(ar0 + ko + 8);
                A1.q[0] = *(const uint4*)(ar1 + ko);
                A1.q[1] = *(const uint4*)(ar1 + ko + 8);
                #pragma unroll
                for (int c = 0; c < 4; ++c) {
                    int ct = ctg * 4 + c;
                    Frag b;
                    const uint32_t* bp = Wsm + ((size_t)(kt * 16 + ct) * 32 + lane) * 8;
                    b.q[0] = *(const uint4*)bp;
                    b.q[1] = *(const uint4*)(bp + 4);
                    acc[0][c] = __builtin_amdgcn_wmma_f32_16x16x32_bf16(
                            false, A0.v, false, b.v, (short)0, acc[0][c], false, false);
                    acc[1][c] = __builtin_amdgcn_wmma_f32_16x16x32_bf16(
                            false, A1.v, false, b.v, (short)0, acc[1][c], false, false);
                }
            }
            #pragma unroll
            for (int rr = 0; rr < 2; ++rr) {
                #pragma unroll
                for (int c = 0; c < 4; ++c) {
                    int ct = ctg * 4 + c;
                    float bias = b1s[ct * 16 + m];
                    #pragma unroll
                    for (int j = 0; j < 8; ++j) {
                        float v = acc[rr][c][j] + bias;
                        v = v > 0.0f ? v : 0.0f;
                        Hf[((rtg * 2 + rr) * 16 + j + half * 8) * HFS + ct * 16 + m] = v;
                    }
                }
            }
        }
        __syncthreads();

        // ---- LayerNorm (fp32) -> bf16 hidden; prefetch next block's gather rows ----
        {
            const float* hr = Hf + r4 * HFS + p4 * 64;
            float s = 0.f, s2 = 0.f;
            #pragma unroll 8
            for (int i = 0; i < 64; ++i) { float v = hr[i]; s += v; s2 += v * v; }
            s  += __shfl_xor(s, 1);  s  += __shfl_xor(s, 2);
            s2 += __shfl_xor(s2, 1); s2 += __shfl_xor(s2, 2);
            float mean = s * (1.0f / 256.0f);
            float var  = s2 * (1.0f / 256.0f) - mean * mean;
            float rstd = __frsqrt_rn(var + 1e-5f);

            if (gnext >= 0 && p4 < 3)      // warm L2/L0 for next gather (312B = 3 lines)
                __builtin_prefetch(var_grid + (size_t)gnext * C_IN + p4 * 32, 0, 3);

            unsigned short* dr = Hbf + r4 * HBS + p4 * 64;
            #pragma unroll 8
            for (int i = 0; i < 64; ++i) {
                int cc = p4 * 64 + i;
                float v = (hr[i] - mean) * rstd * lnSs[cc] + lnOs[cc];
                dr[i] = f2bf(v);
            }
        }
        __syncthreads();

        // ---- GEMM2: out = hidden_bf @ W2 + b2; 2x4 tiles per wave, streaming stores ----
        {
            const unsigned short* hr0 = Hbf + ((rtg * 2 + 0) * 16 + m) * HBS;
            const unsigned short* hr1 = Hbf + ((rtg * 2 + 1) * 16 + m) * HBS;
            const uint32_t* W2sm = Wsm + W1F_DW;
            v8f acc[2][4] = {};
            #pragma unroll
            for (int kt = 0; kt < 8; ++kt) {
                int ko = kt * 32 + half * 16;
                Frag A0, A1;
                A0.q[0] = *(const uint4*)(hr0 + ko);
                A0.q[1] = *(const uint4*)(hr0 + ko + 8);
                A1.q[0] = *(const uint4*)(hr1 + ko);
                A1.q[1] = *(const uint4*)(hr1 + ko + 8);
                #pragma unroll
                for (int c = 0; c < 4; ++c) {
                    int ct = ctg * 4 + c;
                    Frag b;
                    const uint32_t* bp = W2sm + ((size_t)(kt * 16 + ct) * 32 + lane) * 8;
                    b.q[0] = *(const uint4*)bp;
                    b.q[1] = *(const uint4*)(bp + 4);
                    acc[0][c] = __builtin_amdgcn_wmma_f32_16x16x32_bf16(
                            false, A0.v, false, b.v, (short)0, acc[0][c], false, false);
                    acc[1][c] = __builtin_amdgcn_wmma_f32_16x16x32_bf16(
                            false, A1.v, false, b.v, (short)0, acc[1][c], false, false);
                }
            }
            size_t rowbase = (size_t)blk * RB;
            #pragma unroll
            for (int rr = 0; rr < 2; ++rr) {
                size_t rb = rowbase + (rtg * 2 + rr) * 16 + half * 8;
                #pragma unroll
                for (int c = 0; c < 4; ++c) {
                    int ct = ctg * 4 + c;
                    float bias = b2s[ct * 16 + m];
                    #pragma unroll
                    for (int j = 0; j < 8; ++j) {
                        __builtin_nontemporal_store(acc[rr][c][j] + bias,
                            &out[(rb + j) * DM + ct * 16 + m]);
                    }
                }
            }
        }
        // no extra barrier needed: next gather only writes Abf (unused by GEMM2),
        // and the post-gather barrier orders all other cross-phase LDS hazards.
    }
}

extern "C" void kernel_launch(void* const* d_in, const int* in_sizes, int n_in,
                              void* d_out, int out_size, void* d_ws, size_t ws_size,
                              hipStream_t stream) {
    const float* var_grid   = (const float*)d_in[0];
    const int*   nearby_idx = (const int*)d_in[1];
    const float* means      = (const float*)d_in[2];
    const float* stds       = (const float*)d_in[3];
    const float* W1         = (const float*)d_in[4];
    const float* b1         = (const float*)d_in[5];
    const float* ln_scale   = (const float*)d_in[6];
    const float* ln_offset  = (const float*)d_in[7];
    const float* W2         = (const float*)d_in[8];
    const float* b2         = (const float*)d_in[9];
    float* out = (float*)d_out;

    uint32_t* wfrag = (uint32_t*)d_ws;                 // W1F | W2F contiguous
    float*    b1f   = (float*)((uint32_t*)d_ws + B1F_DW_OFF);

    prep_b1<<<1, 256, 0, stream>>>(means, stds, W1, b1, b1f);
    prep_w1frag<<<W1F_DW / 256, 256, 0, stream>>>(stds, W1, wfrag);
    prep_w2frag<<<W2F_DW / 256, 256, 0, stream>>>(W2, wfrag + W1F_DW);

    int grid = NBLK < PERSIST ? NBLK : PERSIST;
    weather_encoder_main<<<grid, 256, SM_TOTAL, stream>>>(
        var_grid, nearby_idx, wfrag, b1f, ln_scale, ln_offset, b2, out);
}
